// GatedGCN_56770877718753
// MI455X (gfx1250) — matrix operand order repaired and checked by reading, not compiled, not run on hardware
//
#include <hip/hip_runtime.h>
#include <hip/hip_bf16.h>

// ---------------- problem dimensions ----------------
#define TT    30
#define BB    512
#define HH    256
#define UEU   10
#define PEP   16
#define SS    282      // H + UE + PE
#define SP    288      // S / (H+UE) padded to multiple of 32 (WMMA bf16 K-step)
#define G3S   846      // 3*S
#define G3SP  848      // padded to multiple of 16 (WMMA N tile)
#define G3H   768      // 3*H
#define NB    15360    // B*T nodes (also T*B rows)
#define EE    122880   // edges
#define DEG   8
#define OUTH  256

typedef unsigned short u16;
typedef __attribute__((ext_vector_type(16))) __bf16 v16bf;
typedef __attribute__((ext_vector_type(8)))  float  v8f;

union Frag8 { unsigned u[8]; v16bf v; };

__device__ __forceinline__ u16 f2bf(float f) {
  union { float f; unsigned u; } v; v.f = f;
  unsigned r = v.u + 0x7FFFu + ((v.u >> 16) & 1u);   // round-to-nearest-even
  return (u16)(r >> 16);
}
__device__ __forceinline__ float bf2f(u16 h) {
  union { unsigned u; float f; } v; v.u = ((unsigned)h) << 16;
  return v.f;
}
__device__ __forceinline__ float sigf(float x) {
  return 1.0f / (1.0f + __expf(-x));
}

// ---------------- bf16 WMMA GEMM: C[M,Npad] = A[M,K] * W[Npad,K]^T ----------------
// A, W row-major bf16 (K zero-padded). Each wave computes a 64x16 output strip:
// 4 accumulators reuse ONE B (weight) fragment per K-step -> 4 WMMAs per 10 b128
// loads, 4x less weight traffic. 8 waves/block -> block covers 512 rows x 16 cols.
__global__ void k_gemm_bf16(const u16* __restrict__ A, const u16* __restrict__ W,
                            float* __restrict__ C,
                            int lda, int ldb, int ldc, int K, int nvalid,
                            const float* __restrict__ bias, int act)
{
  const int lane = threadIdx.x & 31;
  const int wave = threadIdx.x >> 5;
  const int mBase = blockIdx.y * 512 + wave * 64;
  const int nBase = blockIdx.x * 16;
  const int half = lane >> 4;     // 0 or 1
  const int l16  = lane & 15;
  long aRow[4];
#pragma unroll
  for (int i = 0; i < 4; ++i) aRow[i] = (long)(mBase + i * 16 + l16) * lda;
  const long wRow = (long)(nBase + l16) * ldb;

  v8f acc0 = {0.f,0.f,0.f,0.f,0.f,0.f,0.f,0.f};
  v8f acc1 = acc0, acc2 = acc0, acc3 = acc0;

  for (int k0 = 0; k0 < K; k0 += 32) {
    Frag8 b;
#pragma unroll
    for (int v = 0; v < 8; ++v) {
      // B 32x16 bf16 layout (same K order as A): lane=N, VGPR v holds K = (v&4?16:0)+half*8+(v&3)*2, +1
      b.u[v] = *(const unsigned*)(W + wRow + k0 + ((v & 4) ? 16 : 0) + half * 8 + (v & 3) * 2);
    }
    Frag8 a0, a1, a2, a3;
#pragma unroll
    for (int v = 0; v < 8; ++v) {
      // A 16x32 bf16 layout (ISA 7.12.2): VGPR v holds K = (v&4?16:0)+half*8+(v&3)*2, +1
      int ka = k0 + ((v & 4) ? 16 : 0) + half * 8 + (v & 3) * 2;
      a0.u[v] = *(const unsigned*)(A + aRow[0] + ka);
      a1.u[v] = *(const unsigned*)(A + aRow[1] + ka);
      a2.u[v] = *(const unsigned*)(A + aRow[2] + ka);
      a3.u[v] = *(const unsigned*)(A + aRow[3] + ka);
    }
    acc0 = __builtin_amdgcn_wmma_f32_16x16x32_bf16(false, a0.v, false, b.v, (short)0, acc0, false, false);
    acc1 = __builtin_amdgcn_wmma_f32_16x16x32_bf16(false, a1.v, false, b.v, (short)0, acc1, false, false);
    acc2 = __builtin_amdgcn_wmma_f32_16x16x32_bf16(false, a2.v, false, b.v, (short)0, acc2, false, false);
    acc3 = __builtin_amdgcn_wmma_f32_16x16x32_bf16(false, a3.v, false, b.v, (short)0, acc3, false, false);
  }

  const int n = nBase + l16;
  if (n < nvalid) {
    float bv = bias ? bias[n] : 0.0f;
    v8f* accs[4] = { &acc0, &acc1, &acc2, &acc3 };
#pragma unroll
    for (int i = 0; i < 4; ++i) {
#pragma unroll
      for (int r = 0; r < 8; ++r) {
        int m = mBase + i * 16 + half * 8 + r;   // C layout: VGPR r -> M = r + half*8
        float val = (*accs[i])[r] + bv;
        if (act) val = tanhf(val);
        C[(long)m * ldc + n] = val;
      }
    }
  }
}

// ---------------- helpers ----------------
__global__ void k_zero32(float* __restrict__ p, long n) {
  long i = (long)blockIdx.x * blockDim.x + threadIdx.x;
  if (i < n) p[i] = 0.0f;
}

__global__ void k_cvt_weight(const float* __restrict__ src, u16* __restrict__ dst,
                             int rows, int K, int rowsPad, int Kpad)
{
  long i = (long)blockIdx.x * blockDim.x + threadIdx.x;
  long tot = (long)rowsPad * Kpad;
  if (i >= tot) return;
  int r = (int)(i / Kpad), k = (int)(i % Kpad);
  float v = (r < rows && k < K) ? src[(long)r * K + k] : 0.0f;
  dst[i] = f2bf(v);
}

// inp[row t*B+b] = concat(utter_hidden, ub) padded to 288, bf16
__global__ void k_pack_inp(const float* __restrict__ uh, const float* __restrict__ ub,
                           u16* __restrict__ inp)
{
  long i = (long)blockIdx.x * blockDim.x + threadIdx.x;
  if (i >= (long)NB * SP) return;
  int row = (int)(i / SP), c = (int)(i % SP);
  float v;
  if (c < HH)            v = uh[(long)row * HH + c];
  else if (c < HH + UEU) v = ub[(long)row * UEU + (c - HH)];
  else                   v = 0.0f;
  inp[i] = f2bf(v);
}

// one GRU step (gates, torch order r,z,n); writes h_next (f32+bf16) and ys slice
__global__ void k_gru_gate(const float* __restrict__ gx, const float* __restrict__ gh,
                           const float* __restrict__ bih, const float* __restrict__ bhh,
                           const float* __restrict__ hcur,
                           float* __restrict__ hnext, u16* __restrict__ hbf,
                           float* __restrict__ ys)
{
  int idx = blockIdx.x * blockDim.x + threadIdx.x;
  if (idx >= BB * HH) return;
  int b = idx >> 8, c = idx & 255;
  const float* gxr = gx + (long)b * G3H;
  const float* ghr = gh + (long)b * G3H;
  float xr = gxr[c]        + bih[c];
  float xz = gxr[HH + c]   + bih[HH + c];
  float xn = gxr[2*HH + c] + bih[2*HH + c];
  float hr = ghr[c]        + bhh[c];
  float hz = ghr[HH + c]   + bhh[HH + c];
  float hn = ghr[2*HH + c] + bhh[2*HH + c];
  float r = sigf(xr + hr);
  float z = sigf(xz + hz);
  float nn = tanhf(xn + r * hn);
  float h  = hcur[idx];
  float hnew = (1.0f - z) * nn + z * h;
  hnext[idx] = hnew;
  hbf[idx]   = f2bf(hnew);
  ys[idx]    = hnew;
}

// cat_bf[row, 0:256]=ysF, [256:512]=ysB  (bf16)
__global__ void k_cat_rnn(const float* __restrict__ ysF, const float* __restrict__ ysB,
                          u16* __restrict__ cat)
{
  long i = (long)blockIdx.x * blockDim.x + threadIdx.x;
  if (i >= (long)NB * 512) return;
  int row = (int)(i >> 9), c = (int)(i & 511);
  float v = (c < HH) ? ysF[(long)row * HH + c] : ysB[(long)row * HH + (c - HH)];
  cat[i] = f2bf(v);
}

// node n = b*T + t ; xin[n] = concat(rnnx[t*B+b], posemb[t], ub.flat[n]) padded, bf16
__global__ void k_build_xin(const float* __restrict__ rnnx, const float* __restrict__ pos,
                            const float* __restrict__ ub, u16* __restrict__ xin)
{
  long i = (long)blockIdx.x * blockDim.x + threadIdx.x;
  if (i >= (long)NB * SP) return;
  int n = (int)(i / SP), c = (int)(i % SP);
  int t = n % TT, b = n / TT;
  float v;
  if (c < HH)             v = rnnx[((long)t * BB + b) * HH + c];
  else if (c < HH + PEP)  v = pos[t * PEP + (c - HH)];
  else if (c < SS)        v = ub[(long)n * UEU + (c - HH - PEP)];   // faithful turn-major flat
  else                    v = 0.0f;
  xin[i] = f2bf(v);
}

// per-node edge gates + aggregation (edge list is dst-major: node n owns e in [n*8, n*8+8))
__global__ void k_edge(const float* __restrict__ Gx, const float* __restrict__ Gh,
                       const float* __restrict__ bih, const float* __restrict__ bhh,
                       const u16* __restrict__ xin,
                       const int* __restrict__ src, const int* __restrict__ dst,
                       const float* __restrict__ ew, u16* __restrict__ aggr)
{
  int n = blockIdx.x;
  int c = threadIdx.x;                      // 288 threads
  if (c >= SS) { aggr[(long)n * SP + c] = 0; return; }
  long e0 = (long)n * DEG;
  int dn = dst[e0];                         // constant over the node's edge block
  const float* ghr = Gh + (long)dn * G3SP;
  float hr = ghr[c]        + bhh[c];
  float hz = ghr[SS + c]   + bhh[SS + c];
  float hn = ghr[2*SS + c] + bhh[2*SS + c];
  float h  = bf2f(xin[(long)dn * SP + c]);
  float acc = 0.0f;
  for (int d = 0; d < DEG; ++d) {
    long e = e0 + d;
    int s  = src[e];
    float w = ew[e];
    const float* gxr = Gx + (long)s * G3SP;
    float xr = gxr[c]        + bih[c];
    float xz = gxr[SS + c]   + bih[SS + c];
    float xn = gxr[2*SS + c] + bih[2*SS + c];
    float r  = sigf(xr + hr);
    float z  = sigf(xz + hz);
    float nn = tanhf(xn + r * hn);
    acc += w * ((1.0f - z) * nn + z * h);
  }
  aggr[(long)n * SP + c] = f2bf(acc);
}

// cell2 gates: h2 = GRUCell2(aggr, xin)  (Ax = aggr*Wih^T, Ah = xin*Whh^T already done)
__global__ void k_cell2(const float* __restrict__ Ax, const float* __restrict__ Ah,
                        const float* __restrict__ bih, const float* __restrict__ bhh,
                        const u16* __restrict__ xin, u16* __restrict__ h2)
{
  int n = blockIdx.x;
  int c = threadIdx.x;
  if (c >= SS) { h2[(long)n * SP + c] = 0; return; }
  const float* ax = Ax + (long)n * G3SP;
  const float* ah = Ah + (long)n * G3SP;
  float r  = sigf(ax[c] + bih[c] + ah[c] + bhh[c]);
  float z  = sigf(ax[SS + c] + bih[SS + c] + ah[SS + c] + bhh[SS + c]);
  float nn = tanhf(ax[2*SS + c] + bih[2*SS + c] + r * (ah[2*SS + c] + bhh[2*SS + c]));
  float h  = bf2f(xin[(long)n * SP + c]);
  h2[(long)n * SP + c] = f2bf((1.0f - z) * nn + z * h);
}

// xk (already tanh'd by GEMM epilogue): xsum += xk ; xin[:,0:256] = bf16(xk)
__global__ void k_combine(const float* __restrict__ xk, float* __restrict__ xsum,
                          u16* __restrict__ xin)
{
  long i = (long)blockIdx.x * blockDim.x + threadIdx.x;
  if (i >= (long)NB * HH) return;
  int n = (int)(i >> 8), c = (int)(i & 255);
  float v = xk[i];
  xsum[i] += v;
  xin[(long)n * SP + c] = f2bf(v);
}

// xs = LayerNorm(tanh(xsum)); cat2[row t*B+b] = concat(bf16(rnnx[row]), bf16(xs[node]))
__global__ void k_ln_cat2(const float* __restrict__ xsum, const float* __restrict__ rnnx,
                          const float* __restrict__ lng, const float* __restrict__ lnb,
                          u16* __restrict__ cat)
{
  __shared__ float red[HH];
  int n = blockIdx.x, c = threadIdx.x;        // 256 threads
  float v = tanhf(xsum[(long)n * HH + c]);
  red[c] = v; __syncthreads();
  for (int s = 128; s > 0; s >>= 1) { if (c < s) red[c] += red[c + s]; __syncthreads(); }
  float mu = red[0] * (1.0f / HH);
  __syncthreads();
  float d = v - mu;
  red[c] = d * d; __syncthreads();
  for (int s = 128; s > 0; s >>= 1) { if (c < s) red[c] += red[c + s]; __syncthreads(); }
  float var = red[0] * (1.0f / HH);
  float xs = lng[c] * d * rsqrtf(var + 1e-5f) + lnb[c];
  int t = n % TT, b = n / TT;
  long row = (long)t * BB + b;
  cat[row * 512 + HH + c] = f2bf(xs);
  cat[row * 512 + c]      = f2bf(rnnx[row * HH + c]);
}

// ---------------- host orchestration ----------------
extern "C" void kernel_launch(void* const* d_in, const int* in_sizes, int n_in,
                              void* d_out, int out_size, void* d_ws, size_t ws_size,
                              hipStream_t stream)
{
  (void)in_sizes; (void)n_in; (void)out_size; (void)ws_size;

  const float* uh      = (const float*)d_in[0];
  const float* ub      = (const float*)d_in[1];
  const float* ew      = (const float*)d_in[2];
  const float* wih_f   = (const float*)d_in[3];
  const float* whh_f   = (const float*)d_in[4];
  const float* bih_f   = (const float*)d_in[5];
  const float* bhh_f   = (const float*)d_in[6];
  const float* wih_b   = (const float*)d_in[7];
  const float* whh_b   = (const float*)d_in[8];
  const float* bih_b   = (const float*)d_in[9];
  const float* bhh_b   = (const float*)d_in[10];
  const float* lin1_w  = (const float*)d_in[11];
  const float* lin1_b  = (const float*)d_in[12];
  const float* posemb  = (const float*)d_in[13];
  const float* c1_wih  = (const float*)d_in[14];
  const float* c1_whh  = (const float*)d_in[15];
  const float* c1_bih  = (const float*)d_in[16];
  const float* c1_bhh  = (const float*)d_in[17];
  const float* c2_wih  = (const float*)d_in[18];
  const float* c2_whh  = (const float*)d_in[19];
  const float* c2_bih  = (const float*)d_in[20];
  const float* c2_bhh  = (const float*)d_in[21];
  const float* g_w[3]  = { (const float*)d_in[22], (const float*)d_in[24], (const float*)d_in[26] };
  const float* g_b[3]  = { (const float*)d_in[23], (const float*)d_in[25], (const float*)d_in[27] };
  const float* ln_g    = (const float*)d_in[28];
  const float* ln_b    = (const float*)d_in[29];
  const float* lin2_w  = (const float*)d_in[30];
  const float* lin2_b  = (const float*)d_in[31];
  const int*   eidx    = (const int*)d_in[32];
  const int*   src     = eidx;
  const int*   dst     = eidx + EE;

  char* ws = (char*)d_ws;
  size_t off = 0;
  auto take = [&](size_t bytes) -> char* {
    char* p = ws + off; off = (off + bytes + 255) & ~(size_t)255; return p;
  };

  // persistent bf16 weights
  u16* wihF_bf = (u16*)take((size_t)G3H * SP * 2);
  u16* wihB_bf = (u16*)take((size_t)G3H * SP * 2);
  u16* whhF_bf = (u16*)take((size_t)G3H * HH * 2);
  u16* whhB_bf = (u16*)take((size_t)G3H * HH * 2);
  u16* lin1_bf = (u16*)take((size_t)HH * 512 * 2);
  u16* c1wih_bf = (u16*)take((size_t)G3SP * SP * 2);
  u16* c1whh_bf = (u16*)take((size_t)G3SP * SP * 2);
  u16* c2wih_bf = (u16*)take((size_t)G3SP * SP * 2);
  u16* c2whh_bf = (u16*)take((size_t)G3SP * SP * 2);
  u16* gw_bf[3];
  for (int i = 0; i < 3; ++i) gw_bf[i] = (u16*)take((size_t)HH * SP * 2);
  u16* lin2_bf = (u16*)take((size_t)HH * 512 * 2);

  // persistent activations
  float* ysF  = (float*)take((size_t)NB * HH * 4);
  float* ysB  = (float*)take((size_t)NB * HH * 4);
  float* rnnx = (float*)take((size_t)NB * HH * 4);
  u16*   catb = (u16*)take((size_t)NB * 512 * 2);

  // phase-aliased region R (conv layout sized, GRU layout overlaid at front)
  size_t Rbase = off;
  u16*   xin_bf  = (u16*)take((size_t)NB * SP * 2);
  float* Gxb     = (float*)take((size_t)NB * G3SP * 4);
  float* Ghb     = (float*)take((size_t)NB * G3SP * 4);
  u16*   aggr_bf = (u16*)take((size_t)NB * SP * 2);
  u16*   h2_bf   = (u16*)take((size_t)NB * SP * 2);
  float* tmp     = (float*)take((size_t)NB * HH * 4);
  float* xsum    = (float*)take((size_t)NB * HH * 4);
  // GRU-phase overlay
  size_t a = Rbase;
  auto takeA = [&](size_t bytes) -> char* {
    char* p = ws + a; a = (a + bytes + 255) & ~(size_t)255; return p;
  };
  u16*   inp_bf = (u16*)takeA((size_t)NB * SP * 2);
  float* gxbuf  = (float*)takeA((size_t)NB * G3H * 4);
  float* ghbuf  = (float*)takeA((size_t)BB * G3H * 4);
  float* hA     = (float*)takeA((size_t)BB * HH * 4);
  float* hB     = (float*)takeA((size_t)BB * HH * 4);
  u16*   hbf    = (u16*)takeA((size_t)BB * HH * 2);

  auto gemm = [&](const u16* A, const u16* W, float* C, int M, int Npad, int K,
                  int lda, int ldb, int ldc, int nvalid, const float* bias, int act) {
    dim3 grid(Npad / 16, M / 512);              // block = 8 waves x 64 rows = 512 rows
    k_gemm_bf16<<<grid, 256, 0, stream>>>(A, W, C, lda, ldb, ldc, K, nvalid, bias, act);
  };
  auto cvt = [&](const float* s, u16* d, int rows, int K, int rowsPad, int Kpad) {
    long tot = (long)rowsPad * Kpad;
    k_cvt_weight<<<(unsigned)((tot + 255) / 256), 256, 0, stream>>>(s, d, rows, K, rowsPad, Kpad);
  };
  auto zero = [&](float* p, long n) {
    k_zero32<<<(unsigned)((n + 255) / 256), 256, 0, stream>>>(p, n);
  };

  // ---- weight conversion ----
  cvt(wih_f, wihF_bf, G3H, HH + UEU, G3H, SP);
  cvt(wih_b, wihB_bf, G3H, HH + UEU, G3H, SP);
  cvt(whh_f, whhF_bf, G3H, HH, G3H, HH);
  cvt(whh_b, whhB_bf, G3H, HH, G3H, HH);
  cvt(lin1_w, lin1_bf, HH, 512, HH, 512);
  cvt(c1_wih, c1wih_bf, G3S, SS, G3SP, SP);
  cvt(c1_whh, c1whh_bf, G3S, SS, G3SP, SP);
  cvt(c2_wih, c2wih_bf, G3S, SS, G3SP, SP);
  cvt(c2_whh, c2whh_bf, G3S, SS, G3SP, SP);
  for (int i = 0; i < 3; ++i) cvt(g_w[i], gw_bf[i], HH, SS, HH, SP);
  cvt(lin2_w, lin2_bf, HH, 512, HH, 512);

  // ---- pack GRU input ----
  {
    long tot = (long)NB * SP;
    k_pack_inp<<<(unsigned)((tot + 255) / 256), 256, 0, stream>>>(uh, ub, inp_bf);
  }

  // ---- forward GRU ----
  gemm(inp_bf, wihF_bf, gxbuf, NB, G3H, SP, SP, SP, G3H, G3H, nullptr, 0);
  zero(hA, (long)BB * HH);
  zero((float*)hbf, (long)BB * HH / 2);
  for (int t = 0; t < TT; ++t) {
    gemm(hbf, whhF_bf, ghbuf, BB, G3H, HH, HH, HH, G3H, G3H, nullptr, 0);
    k_gru_gate<<<(BB * HH + 255) / 256, 256, 0, stream>>>(
        gxbuf + (long)t * BB * G3H, ghbuf, bih_f, bhh_f, hA, hB, hbf,
        ysF + (long)t * BB * HH);
    float* tsw = hA; hA = hB; hB = tsw;
  }

  // ---- backward GRU (reuse gxbuf) ----
  gemm(inp_bf, wihB_bf, gxbuf, NB, G3H, SP, SP, SP, G3H, G3H, nullptr, 0);
  zero(hA, (long)BB * HH);
  zero((float*)hbf, (long)BB * HH / 2);
  for (int t = TT - 1; t >= 0; --t) {
    gemm(hbf, whhB_bf, ghbuf, BB, G3H, HH, HH, HH, G3H, G3H, nullptr, 0);
    k_gru_gate<<<(BB * HH + 255) / 256, 256, 0, stream>>>(
        gxbuf + (long)t * BB * G3H, ghbuf, bih_b, bhh_b, hA, hB, hbf,
        ysB + (long)t * BB * HH);
    float* tsw = hA; hA = hB; hB = tsw;
  }

  // ---- lin1: rnn_x = tanh([ysF,ysB] @ lin1_w^T + b) ----
  {
    long tot = (long)NB * 512;
    k_cat_rnn<<<(unsigned)((tot + 255) / 256), 256, 0, stream>>>(ysF, ysB, catb);
  }
  gemm(catb, lin1_bf, rnnx, NB, HH, 512, 512, 512, HH, HH, lin1_b, 1);

  // ---- graph convs (GRU-phase buffers are dead now; conv layout takes over R) ----
  {
    long tot = (long)NB * SP;
    k_build_xin<<<(unsigned)((tot + 255) / 256), 256, 0, stream>>>(rnnx, posemb, ub, xin_bf);
  }
  zero(xsum, (long)NB * HH);

  for (int conv = 0; conv < 3; ++conv) {
    // node-level factoring of the edge GRUCell:
    gemm(xin_bf, c1wih_bf, Gxb, NB, G3SP, SP, SP, SP, G3SP, G3S, nullptr, 0);  // Gx = x_src part
    gemm(xin_bf, c1whh_bf, Ghb, NB, G3SP, SP, SP, SP, G3SP, G3S, nullptr, 0);  // Gh = x_dst part
    k_edge<<<NB, SP, 0, stream>>>(Gxb, Ghb, c1_bih, c1_bhh, xin_bf, src, dst, ew, aggr_bf);
    // cell2
    gemm(aggr_bf, c2wih_bf, Gxb, NB, G3SP, SP, SP, SP, G3SP, G3S, nullptr, 0); // Ax
    gemm(xin_bf,  c2whh_bf, Ghb, NB, G3SP, SP, SP, SP, G3SP, G3S, nullptr, 0); // Ah
    k_cell2<<<NB, SP, 0, stream>>>(Gxb, Ghb, c2_bih, c2_bhh, xin_bf, h2_bf);
    // per-conv output linear, fused bias+tanh
    gemm(h2_bf, gw_bf[conv], tmp, NB, HH, SP, SP, SP, HH, HH, g_b[conv], 1);
    k_combine<<<(unsigned)(((long)NB * HH + 255) / 256), 256, 0, stream>>>(tmp, xsum, xin_bf);
  }

  // ---- tanh-sum + LayerNorm + build lin2 input ----
  k_ln_cat2<<<NB, HH, 0, stream>>>(xsum, rnnx, ln_g, ln_b, catb);

  // ---- lin2: out = tanh([rnn_x, gx] @ lin2_w^T + b) -> d_out [T,B,256] ----
  gemm(catb, lin2_bf, (float*)d_out, NB, HH, 512, 512, 512, HH, HH, lin2_b, 1);
}
